// ReactivePolicy_16149077033017
// MI455X (gfx1250) — matrix operand order repaired and checked by reading, not compiled, not run on hardware
//
#include <hip/hip_runtime.h>
#include <math.h>

// ---------------------------------------------------------------------------
// MI455X reasoning:
//  * Dominant cost = streaming features (28MB) twice (stats + GEMM A-stage):
//    ~56MB / 23.3 TB/s ~= 2.4us.  After folding the x10 tile + BatchNorm into
//    an effective 106x256 weight, GEMM is ~4.3 GFLOP in f16 WMMA -> negligible.
//  * A-tiles staged in LDS pre-swizzled to the 16x32 f16 WMMA A layout;
//    Weff pre-packed into B-fragment order in global (64KB, L2-resident),
//    so each lane reads one contiguous 32B chunk per fragment (coalesced).
//  * Epilogue fully fused: beff + LeakyReLU + dot(W2) + cross-lane reduction,
//    h[N,256] never touches memory.
// ---------------------------------------------------------------------------

typedef __attribute__((ext_vector_type(16))) _Float16 v16h;
typedef __attribute__((ext_vector_type(8)))  float    v8f;

#define NCOLS    107        // 1 + FEAT
#define FEAT     106
#define REPEATS  10
#define D_DIM    1060
#define H_DIM    256
#define KPAD     128
#define BATCH    64
#define BN_EPS   1e-5f
#define SLOPE    0.01f

// workspace byte offsets
#define WS_COLSUM   0                   // 128 f32
#define WS_COLSQ    512                 // 128 f32
#define WS_MEAN     1024                // 128 f32
#define WS_RS       1536                // 128 f32
#define WS_BEFF     2048                // 256 f32
#define WS_STARTS   3072                // 65  i32
#define WS_BFRAG    4096                // 16*4*32*16 f16 = 64KB
#define WS_LOGITS   (4096 + 65536)      // N f32

// ---------------------------------------------------------------- zero stats
__global__ void rp_zero_stats(float* colsum, float* colsq) {
    int t = threadIdx.x;
    if (t < 128) { colsum[t] = 0.f; colsq[t] = 0.f; }
}

// --------------------------------------------------- per-column sum / sumsq
// 128 blocks x 256 thr; thread = (half, col); coalesced 106-wide row reads.
__global__ void rp_stats(const float* __restrict__ feat, int N,
                         float* colsum, float* colsq) {
    int tid  = threadIdx.x;
    int col  = tid & 127;
    int half = tid >> 7;
    if (col >= FEAT) return;
    float s = 0.f, q = 0.f;
    for (int r = blockIdx.x * 2 + half; r < N; r += 256) {
        float v = feat[(size_t)r * NCOLS + 1 + col];
        s += v; q += v * v;
    }
    atomicAdd(&colsum[col], s);
    atomicAdd(&colsq[col],  q);
}

__global__ void rp_finalize_stats(const float* colsum, const float* colsq,
                                  int N, float* meanp, float* rsp) {
    int c = threadIdx.x;           // 128 threads
    float mean = 0.f, rs = 0.f;
    if (c < FEAT) {
        float inv = 1.0f / (float)N;
        mean = colsum[c] * inv;
        float var = colsq[c] * inv - mean * mean;
        rs = rsqrtf(var + BN_EPS);
    }
    meanp[c] = mean; rsp[c] = rs;
}

// ---------------------------------------------- beff[h] = b1 + sum beta*W1
__global__ void rp_beff(const float* __restrict__ beta,
                        const float* __restrict__ W1,
                        const float* __restrict__ b1, float* beff) {
    int h = threadIdx.x;           // 256 threads
    float s = b1[h];
    for (int idx = 0; idx < D_DIM; ++idx)
        s += beta[idx] * W1[(size_t)idx * H_DIM + h];
    beff[h] = s;
}

// ------------- Weff (folded gamma*W1 over repeats) packed in B-frag layout
// B fragment (f16, 32x16 per k-step): lane l: n=l&15, hi=l>>4; element p <-> K=(p<8 ? p : 8+p)+8*hi
__global__ void rp_pack_bfrag(const float* __restrict__ gamma,
                              const float* __restrict__ W1,
                              _Float16* __restrict__ bfrag) {
    int j = blockIdx.x;            // 0..127 (K row, padded)
    int h = threadIdx.x;           // 0..255 (output column)
    float w = 0.f;
    if (j < FEAT) {
        for (int r = 0; r < REPEATS; ++r) {
            int row = j + FEAT * r;
            w += gamma[row] * W1[(size_t)row * H_DIM + h];
        }
    }
    int t  = h >> 4, n = h & 15;
    int s  = j >> 5, kl = j & 31;
    int hi = (kl >> 3) & 1, p = (kl & 7) + 8 * (kl >> 4);
    int l  = n + 16 * hi;
    bfrag[(((t * 4 + s) * 32) + l) * 16 + p] = (_Float16)w;
}

// --------------------------------------------------- fused GEMM + epilogue
// grid = N/16 blocks, 256 thr (8 waves). Wave w owns column tiles 2w, 2w+1.
__global__ void rp_gemm(const float* __restrict__ feat,
                        const float* __restrict__ meanp,
                        const float* __restrict__ rsp,
                        const _Float16* __restrict__ bfrag,
                        const float* __restrict__ beff,
                        const float* __restrict__ W2,
                        const float* __restrict__ b2,
                        float* __restrict__ logits) {
    __shared__ __align__(32) _Float16 aF[4][32][16];   // A fragments, 4KB
    __shared__ float sLog[16];

    int tid  = threadIdx.x;
    int lane = tid & 31;
    int wave = tid >> 5;
    int row0 = blockIdx.x * 16;

    if (tid < 16) sLog[tid] = 0.f;

    // Stage A: normalize 16x106 -> f16, pad K to 128, pre-swizzled to frag layout.
    // A layout (f16 16x32): lane=m+16*hi; elem p: K = (p<8 ? p : 8+p) + 8*hi ... inverse below.
#pragma unroll
    for (int e = 0; e < 8; ++e) {
        int lin = tid + 256 * e;          // 0..2047
        int k = lin & 127;                // coalesced along k
        int m = lin >> 7;
        float nv = 0.f;
        if (k < FEAT) {
            float v = feat[(size_t)(row0 + m) * NCOLS + 1 + k];
            nv = (v - meanp[k]) * rsp[k];
        }
        int s   = k >> 5;
        int kk  = k & 31;
        int hi  = (kk >> 3) & 1;
        int kkp = kk - 8 * hi;
        int p   = (kkp < 8) ? kkp : (kkp - 8);
        int l   = m + 16 * hi;
        aF[s][l][p] = (_Float16)nv;
    }
    __syncthreads();

    v8f acc0 = {};
    v8f acc1 = {};
    int t0 = 2 * wave, t1 = 2 * wave + 1;

#pragma unroll
    for (int s = 0; s < 4; ++s) {
        v16h a  = *(const v16h*)&aF[s][lane][0];
        v16h b0 = *(const v16h*)&bfrag[((t0 * 4 + s) * 32 + lane) * 16];
        v16h b1 = *(const v16h*)&bfrag[((t1 * 4 + s) * 32 + lane) * 16];
        acc0 = __builtin_amdgcn_wmma_f32_16x16x32_f16(false, a, false, b0,
                                                      (short)0, acc0, false, false);
        acc1 = __builtin_amdgcn_wmma_f32_16x16x32_f16(false, a, false, b1,
                                                      (short)0, acc1, false, false);
    }

    // Epilogue: h = acc + beff; LeakyReLU; partial logits = h * W2[col]
    int hi = lane >> 4, n = lane & 15;
    float pl[8];
#pragma unroll
    for (int r = 0; r < 8; ++r) pl[r] = 0.f;

    {
        int   c  = 16 * t0 + n;
        float be = beff[c], w2 = W2[c];
#pragma unroll
        for (int r = 0; r < 8; ++r) {
            float hv = acc0[r] + be;
            hv = (hv >= 0.f) ? hv : SLOPE * hv;
            pl[r] += hv * w2;
        }
    }
    {
        int   c  = 16 * t1 + n;
        float be = beff[c], w2 = W2[c];
#pragma unroll
        for (int r = 0; r < 8; ++r) {
            float hv = acc1[r] + be;
            hv = (hv >= 0.f) ? hv : SLOPE * hv;
            pl[r] += hv * w2;
        }
    }

    // reduce over the 16 lanes of each half (xor masks stay inside each half)
#pragma unroll
    for (int off = 8; off >= 1; off >>= 1) {
#pragma unroll
        for (int r = 0; r < 8; ++r) pl[r] += __shfl_xor(pl[r], off, 32);
    }
    if (n == 0) {
#pragma unroll
        for (int r = 0; r < 8; ++r) atomicAdd(&sLog[r + 8 * hi], pl[r]);
    }
    __syncthreads();
    if (tid < 16) logits[row0 + tid] = sLog[tid] + b2[0];
}

// ------------------------------- segment starts (rows are sorted by seg id)
__global__ void rp_starts(const float* __restrict__ feat, int N,
                          int* __restrict__ starts) {
    int i = blockIdx.x * 256 + threadIdx.x;
    if (i >= N) return;
    int s = (int)feat[(size_t)i * NCOLS];
    if (i == 0) {
        starts[s] = 0;
        starts[BATCH] = N;
    } else {
        int sp = (int)feat[(size_t)(i - 1) * NCOLS];
        if (sp != s) starts[s] = i;
    }
}

// ------------------------ segmented log-softmax + scatter into padded grid
__global__ void rp_softmax(const float* __restrict__ logits,
                           const int* __restrict__ starts,
                           float* __restrict__ out, int maxlen) {
    __shared__ float redA[8];
    __shared__ float redB[8];
    __shared__ float bcast;

    int b     = blockIdx.x;
    int tid   = threadIdx.x;
    int lane  = tid & 31;
    int wave  = tid >> 5;
    int start = starts[b];
    int end   = starts[b + 1];
    int cnt   = end - start;

    // max
    float lm = -INFINITY;
    for (int i = start + tid; i < end; i += 256) lm = fmaxf(lm, logits[i]);
#pragma unroll
    for (int off = 16; off >= 1; off >>= 1) lm = fmaxf(lm, __shfl_xor(lm, off, 32));
    if (lane == 0) redA[wave] = lm;
    __syncthreads();
    if (tid == 0) {
        float m = redA[0];
        for (int w = 1; w < 8; ++w) m = fmaxf(m, redA[w]);
        bcast = m;
    }
    __syncthreads();
    float m = bcast;

    // sum of exp
    float ls = 0.f;
    for (int i = start + tid; i < end; i += 256) ls += expf(logits[i] - m);
#pragma unroll
    for (int off = 16; off >= 1; off >>= 1) ls += __shfl_xor(ls, off, 32);
    if (lane == 0) redB[wave] = ls;
    __syncthreads();
    if (tid == 0) {
        float s = 0.f;
        for (int w = 0; w < 8; ++w) s += redB[w];
        bcast = m + logf(s);          // logZ
    }
    __syncthreads();
    float logz = bcast;

    float* orow = out + (size_t)b * maxlen;
    for (int p = tid; p < maxlen; p += 256)
        orow[p] = (p < cnt) ? (logits[start + p] - logz) : -INFINITY;
}

// ---------------------------------------------------------------------------
extern "C" void kernel_launch(void* const* d_in, const int* in_sizes, int n_in,
                              void* d_out, int out_size, void* d_ws, size_t ws_size,
                              hipStream_t stream) {
    const float* feat  = (const float*)d_in[0];
    const float* gamma = (const float*)d_in[1];
    const float* beta  = (const float*)d_in[2];
    const float* W1    = (const float*)d_in[3];
    const float* b1    = (const float*)d_in[4];
    const float* W2    = (const float*)d_in[5];
    const float* b2    = (const float*)d_in[6];
    float* out = (float*)d_out;
    char*  ws  = (char*)d_ws;

    int N      = in_sizes[0] / NCOLS;       // 65536
    int maxlen = out_size / BATCH;          // 2032

    float*    colsum = (float*)(ws + WS_COLSUM);
    float*    colsq  = (float*)(ws + WS_COLSQ);
    float*    meanp  = (float*)(ws + WS_MEAN);
    float*    rsp    = (float*)(ws + WS_RS);
    float*    beff   = (float*)(ws + WS_BEFF);
    int*      starts = (int*)(ws + WS_STARTS);
    _Float16* bfrag  = (_Float16*)(ws + WS_BFRAG);
    float*    logits = (float*)(ws + WS_LOGITS);

    rp_zero_stats<<<1, 256, 0, stream>>>(colsum, colsq);
    rp_stats<<<128, 256, 0, stream>>>(feat, N, colsum, colsq);
    rp_finalize_stats<<<1, 128, 0, stream>>>(colsum, colsq, N, meanp, rsp);
    rp_beff<<<1, 256, 0, stream>>>(beta, W1, b1, beff);
    rp_pack_bfrag<<<KPAD, 256, 0, stream>>>(gamma, W1, bfrag);
    rp_gemm<<<N / 16, 256, 0, stream>>>(feat, meanp, rsp, bfrag, beff, W2, b2, logits);
    rp_starts<<<(N + 255) / 256, 256, 0, stream>>>(feat, N, starts);
    rp_softmax<<<BATCH, 256, 0, stream>>>(logits, starts, out, maxlen);
}
